// SE3Layer_31215822307958
// MI455X (gfx1250) — matrix operand order repaired and checked by reading, not compiled, not run on hardware
//
#include <hip/hip_runtime.h>
#include <hip/hip_bf16.h>

typedef __attribute__((ext_vector_type(16))) _Float16 v16h;
typedef __attribute__((ext_vector_type(8)))  float    v8f;

#define LOG2F_ 0.69314718055994530942f

static __device__ __forceinline__ float sspf(float x) {
  float sp = (x > 20.0f) ? x : log1pf(expf(x));
  return sp - LOG2F_;
}

static __device__ __forceinline__ v8f wmma_f16(v16h a, v16h b, v8f c) {
  return __builtin_amdgcn_wmma_f32_16x16x32_f16(false, a, false, b, (short)0, c,
                                                false, false);
}

// ---- packed-fragment helpers (CDNA5 ISA 7.12.2, wave32) --------------------
// Fragments are stored pre-swizzled: frag blob = 32 lanes x 16 halfs (1024 B).
// Loading one fragment = one 32-byte vector load per lane (2x b128).
static __device__ __forceinline__ v16h load_frag(const _Float16* base, int frag,
                                                 int lane) {
  return *(const v16h*)(base + ((size_t)(frag * 32 + lane) << 4));
}

// inverse map for the 16-bit A layout: value at (row r, k kk) of a 16x32 tile
// lives at lane h*16+r, element j, where kk = h*8+j (j<8) / 16+h*8+(j-8).
static __device__ __forceinline__ int afrag_off(int r, int kk) {
  const int h = (kk < 16) ? (kk >> 3) : ((kk - 16) >> 3);
  const int j = (kk < 16) ? (kk & 7) : (8 + (kk & 7));
  return ((h << 4) + r) * 16 + j;
}

// async global->LDS copy (ASYNCcnt path); lds_off = low 32 bits of generic ptr
static __device__ __forceinline__ void async_gather_b128(unsigned lds_off,
                                                         const void* gaddr) {
  asm volatile("global_load_async_to_lds_b128 %0, %1, off"
               :: "v"(lds_off), "v"(gaddr) : "memory");
}
static __device__ __forceinline__ void wait_async0() {
  asm volatile("s_wait_asynccnt 0x0" ::: "memory");
}

// ===================== kernel 0: pack weights into B fragments ==============
// grid.x = KS*4 fragments, block = 32 lanes. out blob: [frag][lane][16] f16.
__global__ __launch_bounds__(32) void pack_w_kernel(const float* __restrict__ W,
                                                    int krows,
                                                    _Float16* __restrict__ out) {
  const int frag = blockIdx.x, lane = threadIdx.x;
  const int ks = frag >> 2, j = frag & 3;
  const int nn = lane & 15, h = lane >> 4;
  _Float16* o = out + ((size_t)(frag * 32 + lane) << 4);
#pragma unroll
  for (int e = 0; e < 16; ++e) {
    const int k = ks * 32 + ((e < 8) ? (h * 8 + e) : (16 + h * 8 + (e - 8)));
    o[e] = (k < krows) ? (_Float16)W[k * 64 + j * 16 + nn] : (_Float16)0.0f;
  }
}

// ===================== kernel 1: pre_conv = ielin(feat_in, w_pre) -> f16 ====
static __device__ __forceinline__ v16h load_a_f32(const float* p, int ld, int lane) {
  const int r = lane & 15, h = lane >> 4;
  v16h a;
#pragma unroll
  for (int j = 0; j < 16; ++j) {
    const int k = (j < 8) ? (h * 8 + j) : (8 + h * 8 + j);
    a[j] = (_Float16)p[r * ld + k];
  }
  return a;
}

__global__ __launch_bounds__(256) void pre_conv_kernel(
    const float* __restrict__ feat, const _Float16* __restrict__ p_w0,
    const _Float16* __restrict__ p_w1, _Float16* __restrict__ g_pre) {
  const int tid = threadIdx.x, wave = tid >> 5, lane = tid & 31;
  const int row0 = blockIdx.x * 16;
#pragma unroll
  for (int ti = 0; ti < 2; ++ti) {
    const int task = wave + 8 * ti;
    const int comp = task >> 2, j = task & 3;
    const _Float16* Wp = (comp == 0) ? p_w0 : p_w1;
    const int off = (comp == 0) ? 0 : 64 * comp;
    v8f acc = {};
#pragma unroll
    for (int ks = 0; ks < 2; ++ks) {
      v16h a = load_a_f32(feat + (size_t)row0 * 256 + off + ks * 32, 256, lane);
      v16h b = load_frag(Wp, ks * 4 + j, lane);
      acc = wmma_f16(a, b, acc);
    }
    const int h = lane >> 4, nn = lane & 15;
#pragma unroll
    for (int v = 0; v < 8; ++v)
      g_pre[(size_t)(row0 + h * 8 + v) * 256 + off + j * 16 + nn] = (_Float16)acc[v];
  }
}

// ===================== kernel 2: per-node edge pipeline =====================
__global__ __launch_bounds__(256) void edge_kernel(
    const float* __restrict__ xyz, const _Float16* __restrict__ g_pre,
    const _Float16* __restrict__ p_wf1, const _Float16* __restrict__ p_wf2,
    const _Float16* __restrict__ p_wa0, const _Float16* __restrict__ p_wa1,
    const int* __restrict__ src_idx, const int* __restrict__ edge_mask,
    _Float16* __restrict__ g_gatedp) {
  __shared__ __align__(32) _Float16 s_pre[32 * 256];    // gathered pre_conv 16KB
  __shared__ __align__(32) _Float16 s_rbfp[2 * 512];    // rbf, packed A frags
  __shared__ __align__(32) _Float16 s_h1p[2 * 2 * 512]; // ssp(rbf@wf1), packed
  __shared__ __align__(32) _Float16 s_fr[32 * 64];      // radial filter, linear
  __shared__ __align__(32) _Float16 s_c0p[2 * 4 * 512]; // coup0 packed (t,ks)
  __shared__ __align__(32) _Float16 s_c1p[2 * 6 * 512]; // coup1 packed, one m
  __shared__ float s_agg[256];
  __shared__ float s_geo[32 * 8];

  const int n = blockIdx.x;
  const int tid = threadIdx.x, wave = tid >> 5, lane = tid & 31;
  const int t = wave >> 2, jc = wave & 3;

  s_agg[tid] = 0.0f;

  // ---- issue async gather of pre_conv rows first (overlap with geometry) ----
  // each wave moves 4 rows; one wave-wide b128 op = 512B row straight to LDS.
  {
    const unsigned s_pre_off = (unsigned)(uintptr_t)(&s_pre[0]);
#pragma unroll
    for (int q = 0; q < 4; ++q) {
      const int e = wave * 4 + q;
      const int src = src_idx[n * 32 + e];
      async_gather_b128(s_pre_off + (unsigned)(e * 512 + lane * 16),
                        (const char*)g_pre + (size_t)src * 512 + lane * 16);
    }
  }

  // ---- edge geometry ----
  if (tid < 32) {
    const int e = tid;
    const int src = src_idx[n * 32 + e];
    const float rx = xyz[src * 3 + 0] - xyz[n * 3 + 0];
    const float ry = xyz[src * 3 + 1] - xyz[n * 3 + 1];
    const float rz = xyz[src * 3 + 2] - xyz[n * 3 + 2];
    const float d = sqrtf(rx * rx + ry * ry + rz * rz);
    const float dinv = 1.0f / d;
    const float x = d * 0.2f;
    float env = 0.0f;
    if (x < 1.0f) {
      const float x3 = x * x * x;
      env = 1.0f - 10.0f * x3 + 15.0f * x3 * x - 6.0f * x3 * x * x;
    }
    s_geo[e * 8 + 0] = rx * dinv;
    s_geo[e * 8 + 1] = ry * dinv;
    s_geo[e * 8 + 2] = rz * dinv;
    s_geo[e * 8 + 3] = dinv;
    s_geo[e * 8 + 4] = x;
    s_geo[e * 8 + 5] = env;
    s_geo[e * 8 + 6] = (float)edge_mask[n * 32 + e];
  }
  __syncthreads();

  // ---- rbf, written directly in packed A-fragment layout (K padded to 32) --
  for (int idx = tid; idx < 32 * 32; idx += 256) {
    const int e = idx >> 5, k = idx & 31;
    float v = 0.0f;
    if (k < 16)
      v = sinf(s_geo[e * 8 + 4] * (float)k) * s_geo[e * 8 + 3] * s_geo[e * 8 + 5];
    s_rbfp[(e >> 4) * 512 + afrag_off(e & 15, k)] = (_Float16)v;
  }
  wait_async0();           // per-wave: our async rows are in LDS
  __syncthreads();         // all waves' rows + rbf visible

  // ---- h1 = ssp(rbf @ w_f1), stored packed ----
  {
    v16h a = load_frag(s_rbfp, t, lane);
    v16h b = load_frag(p_wf1, jc, lane);
    v8f c = {};
    c = wmma_f16(a, b, c);
    const int h = lane >> 4, nn = lane & 15;
    const int ks = jc >> 1, kk0 = (jc & 1) * 16 + nn;
#pragma unroll
    for (int v = 0; v < 8; ++v)
      s_h1p[(t * 2 + ks) * 512 + afrag_off(h * 8 + v, kk0)] = (_Float16)sspf(c[v]);
  }
  __syncthreads();

  // ---- fr = h1 @ w_f2, stored linear (consumed elementwise) ----
  {
    v8f c = {};
#pragma unroll
    for (int ks = 0; ks < 2; ++ks) {
      v16h a = load_frag(s_h1p, t * 2 + ks, lane);
      v16h b = load_frag(p_wf2, ks * 4 + jc, lane);
      c = wmma_f16(a, b, c);
    }
    const int h = lane >> 4, nn = lane & 15;
#pragma unroll
    for (int v = 0; v < 8; ++v)
      s_fr[(t * 16 + h * 8 + v) * 64 + jc * 16 + nn] = (_Float16)c[v];
  }
  __syncthreads();

  // ---- l=0 coupling, written packed: [pre0*fr | sum_m pre1[m]*rhat_m*fr] ---
  for (int idx = tid; idx < 32 * 64; idx += 256) {
    const int e = idx >> 6, c = idx & 63;
    const float fr = (float)s_fr[e * 64 + c];
    const float p0 = (float)s_pre[e * 256 + c];
    const float py = (float)s_pre[e * 256 + 64 + c];
    const float pz = (float)s_pre[e * 256 + 128 + c];
    const float px = (float)s_pre[e * 256 + 192 + c];
    const float rx = s_geo[e * 8 + 0], ry = s_geo[e * 8 + 1], rz = s_geo[e * 8 + 2];
    const int tt = e >> 4, r = e & 15, ks = c >> 5, kk = c & 31;
    const int ao = afrag_off(r, kk);
    s_c0p[(tt * 4 + ks) * 512 + ao] = (_Float16)(p0 * fr);
    s_c0p[(tt * 4 + 2 + ks) * 512 + ao] = (_Float16)((py * ry + pz * rz + px * rx) * fr);
  }
  __syncthreads();

  // ---- cout0 = coup0 @ w_a0, masked edge-sum ----
  {
    v8f acc = {};
#pragma unroll
    for (int ks = 0; ks < 4; ++ks) {
      v16h a = load_frag(s_c0p, t * 4 + ks, lane);
      v16h b = load_frag(p_wa0, ks * 4 + jc, lane);
      acc = wmma_f16(a, b, acc);
    }
    const int h = lane >> 4, nn = lane & 15;
    float sum = 0.0f;
#pragma unroll
    for (int v = 0; v < 8; ++v)
      sum += acc[v] * s_geo[(t * 16 + h * 8 + v) * 8 + 6];
    sum += __shfl_xor(sum, 16, 32);
    if (lane < 16) atomicAdd(&s_agg[jc * 16 + nn], sum);
  }

  // ---- l=1 coupling per m (rsh order y,z,x), cout1 = coup1 @ w_a1 ----
  for (int m = 0; m < 3; ++m) {
    __syncthreads();
    for (int idx = tid; idx < 32 * 64; idx += 256) {
      const int e = idx >> 6, c = idx & 63;
      const float fr = (float)s_fr[e * 64 + c];
      const float p0 = (float)s_pre[e * 256 + c];
      const float py = (float)s_pre[e * 256 + 64 + c];
      const float pz = (float)s_pre[e * 256 + 128 + c];
      const float px = (float)s_pre[e * 256 + 192 + c];
      const float rx = s_geo[e * 8 + 0], ry = s_geo[e * 8 + 1], rz = s_geo[e * 8 + 2];
      const float rr = (m == 0) ? ry : (m == 1) ? rz : rx;   // rhat_rsh[m]
      const float pm = (m == 0) ? py : (m == 1) ? pz : px;   // pre1[m]
      const float cm = (m == 0) ? (pz * rx - px * rz)
                     : (m == 1) ? (px * ry - py * rx)
                                : (py * rz - pz * ry);       // cross, rsh m
      const int tt = e >> 4, r = e & 15, ks = c >> 5, kk = c & 31;
      const int ao = afrag_off(r, kk);
      s_c1p[(tt * 6 + ks) * 512 + ao] = (_Float16)(p0 * rr * fr);      // 0x1
      s_c1p[(tt * 6 + 2 + ks) * 512 + ao] = (_Float16)(pm * fr);       // 1x0
      s_c1p[(tt * 6 + 4 + ks) * 512 + ao] = (_Float16)(cm * fr);       // 1x1
    }
    __syncthreads();
    v8f acc = {};
#pragma unroll
    for (int ks = 0; ks < 6; ++ks) {
      v16h a = load_frag(s_c1p, t * 6 + ks, lane);
      v16h b = load_frag(p_wa1, ks * 4 + jc, lane);
      acc = wmma_f16(a, b, acc);
    }
    const int h = lane >> 4, nn = lane & 15;
    float sum = 0.0f;
#pragma unroll
    for (int v = 0; v < 8; ++v)
      sum += acc[v] * s_geo[(t * 16 + h * 8 + v) * 8 + 6];
    sum += __shfl_xor(sum, 16, 32);
    if (lane < 16) atomicAdd(&s_agg[64 + m * 64 + jc * 16 + nn], sum);
  }
  __syncthreads();

  // ---- node epilogue: invariants, gate; write gated in packed A layout ----
  if (tid < 64) {
    const int c = tid;
    const float a0 = s_agg[c];
    const float ay = s_agg[64 + c], az = s_agg[128 + c], ax = s_agg[192 + c];
    const float nrm = sqrtf(ay * ay + az * az + ax * ax + 1e-12f);
    const float g0 = sspf(a0), g1 = sspf(nrm);
    const int rowtile = n >> 4, r = n & 15;
    const int ks = c >> 5, kk = c & 31;
    const int ao = afrag_off(r, kk);
    _Float16* gp = g_gatedp + (size_t)rowtile * 8 * 512;
    gp[(0 * 2 + ks) * 512 + ao] = (_Float16)(a0 * g0);
    gp[(1 * 2 + ks) * 512 + ao] = (_Float16)(ay * g1);
    gp[(2 * 2 + ks) * 512 + ao] = (_Float16)(az * g1);
    gp[(3 * 2 + ks) * 512 + ao] = (_Float16)(ax * g1);
  }
}

// ===================== kernel 3: out = feat_in + ielin(gated, w_b) ==========
__global__ __launch_bounds__(256) void out_kernel(
    const _Float16* __restrict__ g_gatedp, const float* __restrict__ feat,
    const _Float16* __restrict__ p_wb0, const _Float16* __restrict__ p_wb1,
    float* __restrict__ out) {
  const int tid = threadIdx.x, wave = tid >> 5, lane = tid & 31;
  const int rowtile = blockIdx.x, row0 = rowtile * 16;
#pragma unroll
  for (int ti = 0; ti < 2; ++ti) {
    const int task = wave + 8 * ti;
    const int comp = task >> 2, j = task & 3;
    const _Float16* Wp = (comp == 0) ? p_wb0 : p_wb1;
    const int off = (comp == 0) ? 0 : 64 * comp;
    v8f acc = {};
#pragma unroll
    for (int ks = 0; ks < 2; ++ks) {
      v16h a = load_frag(g_gatedp, rowtile * 8 + comp * 2 + ks, lane);
      v16h b = load_frag(Wp, ks * 4 + j, lane);
      acc = wmma_f16(a, b, acc);
    }
    const int h = lane >> 4, nn = lane & 15;
#pragma unroll
    for (int v = 0; v < 8; ++v) {
      const size_t oi = (size_t)(row0 + h * 8 + v) * 256 + off + j * 16 + nn;
      out[oi] = feat[oi] + acc[v];
    }
  }
}

extern "C" void kernel_launch(void* const* d_in, const int* in_sizes, int n_in,
                              void* d_out, int out_size, void* d_ws, size_t ws_size,
                              hipStream_t stream) {
  const float* xyz    = (const float*)d_in[0];
  const float* feat   = (const float*)d_in[1];
  const float* w_f1   = (const float*)d_in[2];
  const float* w_f2   = (const float*)d_in[3];
  const float* w_pre0 = (const float*)d_in[4];
  const float* w_pre1 = (const float*)d_in[5];
  const float* w_a0   = (const float*)d_in[6];
  const float* w_a1   = (const float*)d_in[7];
  const float* w_b0   = (const float*)d_in[8];
  const float* w_b1   = (const float*)d_in[9];
  const int* src_idx   = (const int*)d_in[10];
  const int* edge_mask = (const int*)d_in[11];

  const int n_nodes = in_sizes[0] / 3;   // N = 4096

  _Float16* p = (_Float16*)d_ws;
  _Float16* g_pre    = p; p += (size_t)n_nodes * 256;   // N x 256 f16
  _Float16* g_gatedp = p; p += (size_t)n_nodes * 256;   // packed A frags
  _Float16* p_wf1  = p; p += 4 * 512;    // KS=1
  _Float16* p_wf2  = p; p += 8 * 512;    // KS=2
  _Float16* p_wa0  = p; p += 16 * 512;   // KS=4
  _Float16* p_wa1  = p; p += 24 * 512;   // KS=6
  _Float16* p_wpre0 = p; p += 8 * 512;
  _Float16* p_wpre1 = p; p += 8 * 512;
  _Float16* p_wb0  = p; p += 8 * 512;
  _Float16* p_wb1  = p; p += 8 * 512;

  pack_w_kernel<<<4, 32, 0, stream>>>(w_f1, 16, p_wf1);
  pack_w_kernel<<<8, 32, 0, stream>>>(w_f2, 64, p_wf2);
  pack_w_kernel<<<16, 32, 0, stream>>>(w_a0, 128, p_wa0);
  pack_w_kernel<<<24, 32, 0, stream>>>(w_a1, 192, p_wa1);
  pack_w_kernel<<<8, 32, 0, stream>>>(w_pre0, 64, p_wpre0);
  pack_w_kernel<<<8, 32, 0, stream>>>(w_pre1, 64, p_wpre1);
  pack_w_kernel<<<8, 32, 0, stream>>>(w_b0, 64, p_wb0);
  pack_w_kernel<<<8, 32, 0, stream>>>(w_b1, 64, p_wb1);

  pre_conv_kernel<<<n_nodes / 16, 256, 0, stream>>>(feat, p_wpre0, p_wpre1, g_pre);
  edge_kernel<<<n_nodes, 256, 0, stream>>>(xyz, g_pre, p_wf1, p_wf2, p_wa0, p_wa1,
                                           src_idx, edge_mask, g_gatedp);
  out_kernel<<<n_nodes / 16, 256, 0, stream>>>(g_gatedp, feat, p_wb0, p_wb1,
                                               (float*)d_out);
}
